// FaCoRAttention_16552803958969
// MI455X (gfx1250) — matrix operand order repaired and checked by reading, not compiled, not run on hardware
//
#include <hip/hip_runtime.h>

// ---------------------------------------------------------------------------
// FaCoR attention for MI455X (gfx1250, wave32, WMMA bf16 16x16x32)
// B=256, C=512, H=W=7 -> HW=49 (padded to 64 for WMMA tiles), QK=32
// ---------------------------------------------------------------------------

typedef __bf16 bf16_t;
typedef bf16_t bf16x8  __attribute__((ext_vector_type(8)));
typedef bf16_t bf16x16 __attribute__((ext_vector_type(16)));
typedef float  f32x8   __attribute__((ext_vector_type(8)));

constexpr int B_   = 256;
constexpr int C_   = 512;
constexpr int C2_  = 1024;
constexpr int HW_  = 49;
constexpr int NP_  = 64;          // padded spatial
constexpr int S_   = B_ * HW_;    // 12544 (== 784*16 exactly)
constexpr int QK_  = 32;

#define DEV static __device__ __forceinline__

DEV float wave_sum(float v) {
  for (int o = 16; o > 0; o >>= 1) v += __shfl_xor(v, o, 32);
  return v;
}
DEV float wave_max(float v) {
  for (int o = 16; o > 0; o >>= 1) v = fmaxf(v, __shfl_xor(v, o, 32));
  return v;
}
DEV f32x8 zero8() {
  f32x8 z = {0.f, 0.f, 0.f, 0.f, 0.f, 0.f, 0.f, 0.f};
  return z;
}

// A fragment: 16x32 bf16, row-major source, row m = m0 + (lane&15).
// lanes 0-15: K = k0+[0..15]; lanes 16-31: K = k0+16+[0..15] (matches B fragment)
DEV bf16x16 load_a_frag(const bf16_t* __restrict__ base, int m0, int ld, int k0, int lane) {
  int m    = m0 + (lane & 15);
  int half = lane >> 4;
  const bf16_t* r = base + (size_t)m * ld + k0 + half * 16;
  return *(const bf16x16*)(r);
}

// B fragment: 32x16 bf16 from K-transposed storage: srcT[n][k] contiguous in k.
// lanes 0-15 hold K=0..15 for col n, lanes 16-31 hold K=16..31.
DEV bf16x16 load_b_frag(const bf16_t* __restrict__ baseT, int n0, int ld, int k0, int lane) {
  int n  = n0 + (lane & 15);
  int kh = lane >> 4;
  return *(const bf16x16*)(baseT + (size_t)n * ld + k0 + kh * 16);
}

DEV f32x8 wmma_bf16(bf16x16 a, bf16x16 b, f32x8 c) {
  return __builtin_amdgcn_wmma_f32_16x16x32_bf16(false, a, false, b, (short)0, c, false, false);
}

// ---------------------------------------------------------------------------
// Elementwise / small kernels
// ---------------------------------------------------------------------------

__global__ void k_cvt_bf16(const float* __restrict__ src, bf16_t* __restrict__ dst, int n) {
  int i = blockIdx.x * blockDim.x + threadIdx.x;
  if (i < n) dst[i] = (bf16_t)src[i];
}

// inverse channel-L2 norm per (b, spatial)
__global__ void k_invnorm(const float* __restrict__ x, float* __restrict__ invn) {
  int gid  = (blockIdx.x * blockDim.x + threadIdx.x) >> 5;
  int lane = threadIdx.x & 31;
  if (gid >= B_ * HW_) return;
  int b = gid / HW_, m = gid % HW_;
  const float* px = x + (size_t)b * C_ * HW_ + m;
  float s = 0.f;
  for (int c = lane; c < C_; c += 32) {
    float v = px[(size_t)c * HW_];
    s += v * v;
  }
  s = wave_sum(s);
  if (lane == 0) invn[gid] = rsqrtf(s);
}

// pack concat(x1,x2) (optionally normalized) transposed: xt[s][cc], s=b*49+m
__global__ void k_pack_xt(const float* __restrict__ x1, const float* __restrict__ x2,
                          const float* __restrict__ in1, const float* __restrict__ in2,
                          int use_norm, bf16_t* __restrict__ xt) {
  size_t idx = (size_t)blockIdx.x * blockDim.x + threadIdx.x;
  if (idx >= (size_t)S_ * C2_) return;
  int cc = (int)(idx % C2_);
  int s  = (int)(idx / C2_);
  int b = s / HW_, m = s % HW_;
  float v;
  if (cc < C_) v = x1[((size_t)b * C_ + cc) * HW_ + m];
  else         v = x2[((size_t)b * C_ + (cc - C_)) * HW_ + m];
  if (use_norm) v *= (cc < C_ ? in1[s] : in2[s]);
  xt[idx] = (bf16_t)v;
}

// ---------------------------------------------------------------------------
// WMMA GEMM kernels
// ---------------------------------------------------------------------------

// q/k projection: [32 x 1024] * [1024 x 12544], output stored spatial-major
// (padded): outT[b][64][32] bf16, row = spatial, cols = qk channel (+bias).
__global__ void k_gemm_qk(const bf16_t* __restrict__ W, const float* __restrict__ bias,
                          const bf16_t* __restrict__ xt, bf16_t* __restrict__ outT) {
  int wave = (blockIdx.x * blockDim.x + threadIdx.x) >> 5;
  int lane = threadIdx.x & 31;
  const int NT = S_ / 16;  // 784
  int mt = wave / NT, nt = wave % NT;  // grid sized exactly 2*784 waves
  f32x8 acc = zero8();
  for (int k0 = 0; k0 < C2_; k0 += 32) {
    bf16x16 a = load_a_frag(W, mt * 16, C2_, k0, lane);
    bf16x16 b = load_b_frag(xt, nt * 16, C2_, k0, lane);
    acc = wmma_bf16(a, b, acc);
  }
  int s = nt * 16 + (lane & 15);      // always < S_ (12544 = 784*16)
  int b_ = s / HW_, nsp = s % HW_;
  int half = lane >> 4;
  bf16_t* dst = outT + ((size_t)b_ * NP_ + nsp) * QK_;
#pragma unroll
  for (int v = 0; v < 8; ++v) {
    int o = mt * 16 + v + 8 * half;
    dst[o] = (bf16_t)(acc[v] + bias[o]);
  }
}

// V projection: [512 x 512] * [512 x 12544] over columns [coff, coff+512) of xt.
// Output: vbuf[b][512][64] bf16 (spatial padded), +bias. Wave: 64 rows x 16 cols.
__global__ void k_gemm_v(const bf16_t* __restrict__ W, const float* __restrict__ bias,
                         const bf16_t* __restrict__ xt, int coff, bf16_t* __restrict__ vbuf) {
  int wave = (blockIdx.x * blockDim.x + threadIdx.x) >> 5;
  int lane = threadIdx.x & 31;
  const int NT = S_ / 16;              // 784
  int mg = wave / NT, nt = wave % NT;  // grid exactly 8*784 waves
  f32x8 acc[4] = {zero8(), zero8(), zero8(), zero8()};
  for (int k0 = 0; k0 < C_; k0 += 32) {
    bf16x16 bfrag = load_b_frag(xt, nt * 16, C2_, coff + k0, lane);
#pragma unroll
    for (int i = 0; i < 4; ++i) {
      bf16x16 afrag = load_a_frag(W, mg * 64 + i * 16, C_, k0, lane);
      acc[i] = wmma_bf16(afrag, bfrag, acc[i]);
    }
  }
  int s = nt * 16 + (lane & 15);
  int b_ = s / HW_, nsp = s % HW_;
  int half = lane >> 4;
#pragma unroll
  for (int i = 0; i < 4; ++i) {
#pragma unroll
    for (int v = 0; v < 8; ++v) {
      int o = mg * 64 + i * 16 + v + 8 * half;
      vbuf[((size_t)b_ * C_ + o) * NP_ + nsp] = (bf16_t)(acc[i][v] + bias[o]);
    }
  }
}

// energy[b,n,m] = sum_qc qT[b,n,qc] * kT[b,m,qc]; K=32 -> one WMMA per tile.
// Wave: 16 n-rows x 64 m-cols (A frag reused over 4 B tiles). 1024 waves.
__global__ void k_energy(const bf16_t* __restrict__ qT, const bf16_t* __restrict__ kT,
                         float* __restrict__ energy) {
  int wave = (blockIdx.x * blockDim.x + threadIdx.x) >> 5;
  int lane = threadIdx.x & 31;
  int nt = wave & 3, b = wave >> 2;
  const bf16_t* qb = qT + (size_t)b * NP_ * QK_;
  const bf16_t* kb = kT + (size_t)b * NP_ * QK_;
  bf16x16 af = load_a_frag(qb, nt * 16, QK_, 0, lane);
  int half = lane >> 4;
  float* eb = energy + (size_t)b * NP_ * NP_;
#pragma unroll
  for (int mt = 0; mt < 4; ++mt) {
    bf16x16 bf = load_b_frag(kb, mt * 16, QK_, 0, lane);
    f32x8 acc = wmma_bf16(af, bf, zero8());
    int m = mt * 16 + (lane & 15);
#pragma unroll
    for (int v = 0; v < 8; ++v) {
      int n = nt * 16 + v + 8 * half;
      eb[(size_t)n * NP_ + m] = acc[v];
    }
  }
}

// row softmax over valid 49 entries; mode 0: fp32 -> att_map0, mode 1: bf16 -> attB
__global__ void k_softmax(const float* __restrict__ energy, int mode,
                          float* __restrict__ attF, bf16_t* __restrict__ attB) {
  int gid  = (blockIdx.x * blockDim.x + threadIdx.x) >> 5;
  int lane = threadIdx.x & 31;
  if (gid >= B_ * HW_) return;
  int b = gid / HW_, n = gid % HW_;
  const float* row = energy + ((size_t)b * NP_ + n) * NP_;
  int j0 = lane, j1 = lane + 32;
  float v0 = (j0 < HW_) ? row[j0] : -3.4e38f;
  float v1 = (j1 < HW_) ? row[j1] : -3.4e38f;
  float mx = wave_max(fmaxf(v0, v1));
  float e0 = (j0 < HW_) ? expf(v0 - mx) : 0.f;
  float e1 = (j1 < HW_) ? expf(v1 - mx) : 0.f;
  float inv = 1.f / wave_sum(e0 + e1);
  if (mode == 0) {
    float* orow = attF + ((size_t)b * HW_ + n) * HW_;
    if (j0 < HW_) orow[j0] = e0 * inv;
    if (j1 < HW_) orow[j1] = e1 * inv;
  } else {
    bf16_t* orow = attB + ((size_t)b * NP_ + n) * NP_;
    if (j0 < HW_) orow[j0] = (bf16_t)(e0 * inv);
    if (j1 < HW_) orow[j1] = (bf16_t)(e1 * inv);
  }
}

// out[b,c,x] = sum_y v[b,c,y]*att[b,x,y]; directly accumulate spatial sums into
// fsum[b][c] via f32 atomics (only the sum is needed: adaptive avg pool).
// Wave: 64 c-rows x 16 x-cols, K=64 (2 steps; att y-padding is exact zero).
__global__ void k_outbmm(const bf16_t* __restrict__ v1buf, const bf16_t* __restrict__ v2buf,
                         const bf16_t* __restrict__ attB,
                         float* __restrict__ fsum1, float* __restrict__ fsum2) {
  int wave = (blockIdx.x * blockDim.x + threadIdx.x) >> 5;
  int lane = threadIdx.x & 31;
  int xt = wave & 3;  wave >>= 2;
  int mg = wave & 7;  wave >>= 3;
  int b  = wave & 255; wave >>= 8;
  int which = wave;  // grid exactly 2*256*8*4 waves
  const bf16_t* vbase = (which ? v2buf : v1buf) + (size_t)b * C_ * NP_;
  const bf16_t* abase = attB + (size_t)b * NP_ * NP_;
  float* fs = which ? fsum2 : fsum1;
  f32x8 acc[4] = {zero8(), zero8(), zero8(), zero8()};
  for (int k0 = 0; k0 < NP_; k0 += 32) {
    bf16x16 bfrag = load_b_frag(abase, xt * 16, NP_, k0, lane);
#pragma unroll
    for (int i = 0; i < 4; ++i) {
      bf16x16 afrag = load_a_frag(vbase, mg * 64 + i * 16, NP_, k0, lane);
      acc[i] = wmma_bf16(afrag, bfrag, acc[i]);
    }
  }
  int x = xt * 16 + (lane & 15);
  if (x < HW_) {
    int half = lane >> 4;
#pragma unroll
    for (int i = 0; i < 4; ++i)
#pragma unroll
      for (int v = 0; v < 8; ++v) {
        int c = mg * 64 + i * 16 + v + 8 * half;
        atomicAdd(&fs[(size_t)b * C_ + c], acc[i][v]);
      }
  }
}

// ---------------------------------------------------------------------------
// Channel attention + interaction (fp32 VALU, small)
// ---------------------------------------------------------------------------

// xc[b,o] = Wc[o,:] . concat(f1,f2)[b] + bc[o]
__global__ void k_xc(const float* __restrict__ f1, const float* __restrict__ f2,
                     const float* __restrict__ Wc, const float* __restrict__ bc,
                     float* __restrict__ xc) {
  __shared__ float y[C2_];
  int b = blockIdx.x, t = threadIdx.x;
  for (int i = t; i < C_; i += 256) {
    y[i]       = f1[(size_t)b * C_ + i];
    y[C_ + i]  = f2[(size_t)b * C_ + i];
  }
  __syncthreads();
  for (int o = t; o < C_; o += 256) {
    const float* w = Wc + (size_t)o * C2_;
    float s = bc[o];
    for (int j = 0; j < C2_; ++j) s += w[j] * y[j];
    xc[(size_t)b * C_ + o] = s;
  }
}

// rank-1 channel CA: f{1,2}_1[b,i] = softmax_j(xc_i*xc_j) . f{1,2}[b,:] + f{1,2}[b,i]
__global__ void k_chatt(const float* __restrict__ xc, const float* __restrict__ f1,
                        const float* __restrict__ f2, float* __restrict__ o1,
                        float* __restrict__ o2) {
  int gid  = (blockIdx.x * blockDim.x + threadIdx.x) >> 5;
  int lane = threadIdx.x & 31;
  if (gid >= B_ * C_) return;
  int b = gid / C_, i = gid % C_;
  const float* xr = xc + (size_t)b * C_;
  const float* p1 = f1 + (size_t)b * C_;
  const float* p2 = f2 + (size_t)b * C_;
  float xi = xr[i];
  float mx = -3.4e38f;
  for (int j = lane; j < C_; j += 32) mx = fmaxf(mx, xi * xr[j]);
  mx = wave_max(mx);
  float se = 0.f, s1 = 0.f, s2 = 0.f;
  for (int j = lane; j < C_; j += 32) {
    float e = expf(xi * xr[j] - mx);
    se += e; s1 += e * p1[j]; s2 += e * p2[j];
  }
  se = wave_sum(se); s1 = wave_sum(s1); s2 = wave_sum(s2);
  if (lane == 0) {
    float inv = 1.f / se;
    o1[gid] = s1 * inv + p1[i];
    o2[gid] = s2 * inv + p2[i];
  }
}

// f_2[b,c] = (bmm_sum + sum_m x[b,c,m]*invn[b,m]) / 49   (residual + avg pool)
__global__ void k_means(const float* __restrict__ fsum, const float* __restrict__ x,
                        const float* __restrict__ invn, float* __restrict__ out) {
  int idx = blockIdx.x * blockDim.x + threadIdx.x;
  if (idx >= B_ * C_) return;
  int b = idx / C_;
  float s = fsum[idx];
  const float* px = x + (size_t)idx * HW_;
  const float* pn = invn + (size_t)b * HW_;
  for (int m = 0; m < HW_; ++m) s += px[m] * pn[m];
  out[idx] = s * (1.f / HW_);
}

// channel interaction MLP (1024->128 relu ->1024 sigmoid) + blend, per batch
__global__ void k_final(const float* __restrict__ fa1, const float* __restrict__ fa2,
                        const float* __restrict__ Wi1, const float* __restrict__ bi1,
                        const float* __restrict__ Wi2, const float* __restrict__ bi2,
                        float* __restrict__ out) {
  __shared__ float y[C2_];
  __shared__ float h[128];
  __shared__ float w[C2_];
  int b = blockIdx.x, t = threadIdx.x;
  for (int i = t; i < C_; i += 256) {
    y[i]      = fa1[(size_t)b * C_ + i];
    y[C_ + i] = fa2[(size_t)b * C_ + i];
  }
  __syncthreads();
  if (t < 128) {
    const float* wr = Wi1 + (size_t)t * C2_;
    float s = bi1[t];
    for (int j = 0; j < C2_; ++j) s += wr[j] * y[j];
    h[t] = fmaxf(s, 0.f);
  }
  __syncthreads();
  for (int o = t; o < C2_; o += 256) {
    const float* wr = Wi2 + (size_t)o * 128;
    float s = bi2[o];
    for (int j = 0; j < 128; ++j) s += wr[j] * h[j];
    w[o] = 1.f / (1.f + expf(-s));
  }
  __syncthreads();
  for (int o = t; o < C_; o += 256)
    out[(size_t)b * C_ + o] = y[o] * w[o] + y[C_ + o] * w[C_ + o];
}

// ---------------------------------------------------------------------------
// Launch
// ---------------------------------------------------------------------------

extern "C" void kernel_launch(void* const* d_in, const int* in_sizes, int n_in,
                              void* d_out, int out_size, void* d_ws, size_t ws_size,
                              hipStream_t stream) {
  const float* f1_0 = (const float*)d_in[0];
  const float* x1   = (const float*)d_in[1];
  const float* f2_0 = (const float*)d_in[2];
  const float* x2   = (const float*)d_in[3];
  const float* Wq   = (const float*)d_in[4];
  const float* bq   = (const float*)d_in[5];
  const float* Wk   = (const float*)d_in[6];
  const float* bk   = (const float*)d_in[7];
  const float* Wv1  = (const float*)d_in[8];
  const float* bv1  = (const float*)d_in[9];
  const float* Wv2  = (const float*)d_in[10];
  const float* bv2  = (const float*)d_in[11];
  const float* Wc   = (const float*)d_in[12];
  const float* bc   = (const float*)d_in[13];
  const float* Wi1  = (const float*)d_in[14];
  const float* bi1  = (const float*)d_in[15];
  const float* Wi2  = (const float*)d_in[16];
  const float* bi2  = (const float*)d_in[17];

  float* out_f1s = (float*)d_out;
  float* out_f2s = out_f1s + (size_t)B_ * C_;
  float* out_att = out_f2s + (size_t)B_ * C_;

  char* p = (char*)d_ws;
  auto alloc = [&](size_t bytes) -> char* {
    char* r = p;
    p += (bytes + 255) & ~(size_t)255;
    return r;
  };
  bf16_t* wq_bf  = (bf16_t*)alloc((size_t)QK_ * C2_ * 2);
  bf16_t* wk_bf  = (bf16_t*)alloc((size_t)QK_ * C2_ * 2);
  bf16_t* wv1_bf = (bf16_t*)alloc((size_t)C_ * C_ * 2);
  bf16_t* wv2_bf = (bf16_t*)alloc((size_t)C_ * C_ * 2);
  bf16_t* xt     = (bf16_t*)alloc((size_t)S_ * C2_ * 2);
  float*  invn1  = (float*)alloc((size_t)S_ * 4);
  float*  invn2  = (float*)alloc((size_t)S_ * 4);
  bf16_t* qT     = (bf16_t*)alloc((size_t)B_ * NP_ * QK_ * 2);
  bf16_t* kT     = (bf16_t*)alloc((size_t)B_ * NP_ * QK_ * 2);
  float*  energy = (float*)alloc((size_t)B_ * NP_ * NP_ * 4);
  bf16_t* attB   = (bf16_t*)alloc((size_t)B_ * NP_ * NP_ * 2);
  bf16_t* v1buf  = (bf16_t*)alloc((size_t)B_ * C_ * NP_ * 2);
  bf16_t* v2buf  = (bf16_t*)alloc((size_t)B_ * C_ * NP_ * 2);
  float*  fsum1  = (float*)alloc((size_t)B_ * C_ * 4);
  float*  fsum2  = (float*)alloc((size_t)B_ * C_ * 4);
  float*  xc     = (float*)alloc((size_t)B_ * C_ * 4);
  float*  f1_1   = (float*)alloc((size_t)B_ * C_ * 4);
  float*  f2_1   = (float*)alloc((size_t)B_ * C_ * 4);
  float*  f1_2   = (float*)alloc((size_t)B_ * C_ * 4);
  float*  f2_2   = (float*)alloc((size_t)B_ * C_ * 4);

  // zero K-padding of att (exact zeros for out-BMM K=64) and the mean accumulators
  hipMemsetAsync(attB, 0, (size_t)B_ * NP_ * NP_ * 2, stream);
  hipMemsetAsync(fsum1, 0, (size_t)B_ * C_ * 4, stream);
  hipMemsetAsync(fsum2, 0, (size_t)B_ * C_ * 4, stream);

  // weights -> bf16
  k_cvt_bf16<<<(QK_ * C2_ + 255) / 256, 256, 0, stream>>>(Wq, wq_bf, QK_ * C2_);
  k_cvt_bf16<<<(QK_ * C2_ + 255) / 256, 256, 0, stream>>>(Wk, wk_bf, QK_ * C2_);
  k_cvt_bf16<<<(C_ * C_ + 255) / 256, 256, 0, stream>>>(Wv1, wv1_bf, C_ * C_);
  k_cvt_bf16<<<(C_ * C_ + 255) / 256, 256, 0, stream>>>(Wv2, wv2_bf, C_ * C_);

  // channel-L2 inverse norms
  k_invnorm<<<(B_ * HW_) / 4, 128, 0, stream>>>(x1, invn1);
  k_invnorm<<<(B_ * HW_) / 4, 128, 0, stream>>>(x2, invn2);

  const int packBlocks = (int)(((size_t)S_ * C2_ + 255) / 256);

  // ---- pass 1: raw features -> att_map0 ----
  k_pack_xt<<<packBlocks, 256, 0, stream>>>(x1, x2, invn1, invn2, 0, xt);
  k_gemm_qk<<<(2 * (S_ / 16)) / 4, 128, 0, stream>>>(wq_bf, bq, xt, qT);
  k_gemm_qk<<<(2 * (S_ / 16)) / 4, 128, 0, stream>>>(wk_bf, bk, xt, kT);
  k_energy<<<(B_ * 4) / 4, 128, 0, stream>>>(qT, kT, energy);
  k_softmax<<<(B_ * HW_) / 4, 128, 0, stream>>>(energy, 0, out_att, attB);

  // ---- channel cross attention (independent) ----
  k_xc<<<B_, 256, 0, stream>>>(f1_0, f2_0, Wc, bc, xc);
  k_chatt<<<(B_ * C_) / 4, 128, 0, stream>>>(xc, f1_0, f2_0, f1_1, f2_1);

  // ---- pass 2: normalized features -> spatial outputs (mean-pooled) ----
  k_pack_xt<<<packBlocks, 256, 0, stream>>>(x1, x2, invn1, invn2, 1, xt);
  k_gemm_qk<<<(2 * (S_ / 16)) / 4, 128, 0, stream>>>(wq_bf, bq, xt, qT);
  k_gemm_qk<<<(2 * (S_ / 16)) / 4, 128, 0, stream>>>(wk_bf, bk, xt, kT);
  k_gemm_v<<<(8 * (S_ / 16)) / 4, 128, 0, stream>>>(wv1_bf, bv1, xt, 0, v1buf);
  k_gemm_v<<<(8 * (S_ / 16)) / 4, 128, 0, stream>>>(wv2_bf, bv2, xt, C_, v2buf);
  k_energy<<<(B_ * 4) / 4, 128, 0, stream>>>(qT, kT, energy);
  k_softmax<<<(B_ * HW_) / 4, 128, 0, stream>>>(energy, 1, out_att, attB);
  k_outbmm<<<(2 * B_ * 8 * 4) / 4, 128, 0, stream>>>(v1buf, v2buf, attB, fsum1, fsum2);

  // means incl. normalized residual
  k_means<<<(B_ * C_ + 255) / 256, 256, 0, stream>>>(fsum1, x1, invn1, f1_2);
  k_means<<<(B_ * C_ + 255) / 256, 256, 0, stream>>>(fsum2, x2, invn2, f2_2);

  // channel interaction + final blend
  k_final<<<B_, 256, 0, stream>>>(f1_1, f1_2, Wi1, bi1, Wi2, bi2, out_f1s);
  k_final<<<B_, 256, 0, stream>>>(f2_1, f2_2, Wi1, bi1, Wi2, bi2, out_f2s);
}
